// MultiheadAttentionWithExtras_5471788335654
// MI455X (gfx1250) — matrix operand order repaired and checked by reading, not compiled, not run on hardware
//
#include <hip/hip_runtime.h>
#include <hip/hip_bf16.h>

// ---------------------------------------------------------------------------
// Types / WMMA helpers (CDNA5 gfx1250, wave32)
// ---------------------------------------------------------------------------
typedef __bf16 bf16_t;
typedef __attribute__((ext_vector_type(16))) __bf16 v16bf;
typedef __attribute__((ext_vector_type(8)))  float  v8f;
typedef __attribute__((ext_vector_type(4)))  unsigned int v4u;   // 16 bytes

union Frag {
  v16bf bf;
  v4u   q[2];
};

__device__ __forceinline__ bf16_t f2bf(float f) {
  union { float f; unsigned u; } v; v.f = f;
  unsigned r = v.u + 0x7FFFu + ((v.u >> 16) & 1u);   // round-to-nearest-even
  unsigned short h = (unsigned short)(r >> 16);
  bf16_t b; __builtin_memcpy(&b, &h, 2); return b;
}

__device__ __forceinline__ v8f wmma_bf16(v16bf a, v16bf b, v8f c) {
  return __builtin_amdgcn_wmma_f32_16x16x32_bf16(false, a, false, b, (short)0, c,
                                                 false, false);
}

// A fragment: 16x32 (MxK) row-major tile in LDS, stride lda.
// lanes 0-15: M=lane, K=0..15; lanes 16-31: M=lane-16, K=16..31.
// -> one contiguous 32-byte run per lane: [kb, kb+16).
__device__ __forceinline__ v16bf frag_a(const bf16_t* A, int lda) {
  int lane = threadIdx.x & 31;
  const bf16_t* p = A + (lane & 15) * lda + ((lane < 16) ? 0 : 16);
  Frag f;
  f.q[0] = *(const v4u*)p;
  f.q[1] = *(const v4u*)(p + 8);
  return f.bf;
}

// B fragment from an N x K row-major tile (i.e. the operand transposed in LDS).
// lanes 0-15: N=lane, K=0..15; lanes 16-31: N=lane-16, K=16..31.
// -> one contiguous 32-byte run per lane: [kb, kb+16).
__device__ __forceinline__ v16bf frag_bt(const bf16_t* Bt, int ldb) {
  int lane = threadIdx.x & 31;
  const bf16_t* p = Bt + (lane & 15) * ldb + ((lane < 16) ? 0 : 16);
  Frag f;
  f.q[0] = *(const v4u*)p;
  f.q[1] = *(const v4u*)(p + 8);
  return f.bf;
}

// ---------------------------------------------------------------------------
// Elementwise kernels
// ---------------------------------------------------------------------------
__global__ void kern_pack(const float* __restrict__ src, bf16_t* __restrict__ dst, long n) {
  long i = (long)blockIdx.x * 256 + threadIdx.x;
  if (i < n) dst[i] = f2bf(src[i]);
}

// aqak[h,t,s] = sum_d aQ[h,t,s,d]*aK[h,t,s,d]   (65536 rows of 64)
__global__ void kern_aqak(const float* __restrict__ aQ, const float* __restrict__ aK,
                          float* __restrict__ out) {
  int idx = blockIdx.x * 256 + threadIdx.x;
  const float* qa = aQ + (long)idx * 64;
  const float* ka = aK + (long)idx * 64;
  float s = 0.f;
#pragma unroll 8
  for (int j = 0; j < 64; ++j) s += qa[j] * ka[j];
  out[idx] = s;
}

// SiLU then LayerNorm over 256 columns; one wave per row (8 cols/lane).
__global__ void kern_silu_ln(const float* __restrict__ in, bf16_t* __restrict__ out,
                             const float* __restrict__ gm, const float* __restrict__ bt) {
  int row = blockIdx.x * 4 + (threadIdx.x >> 5);
  int lane = threadIdx.x & 31;
  const float* p = in + (long)row * 256 + lane * 8;
  float v[8]; float s = 0.f;
#pragma unroll
  for (int j = 0; j < 8; ++j) {
    float xx = p[j];
    float sv = xx / (1.0f + __expf(-xx));
    v[j] = sv; s += sv;
  }
  for (int o = 16; o > 0; o >>= 1) s += __shfl_xor(s, o, 32);
  float mean = s * (1.0f / 256.0f);
  float q = 0.f;
#pragma unroll
  for (int j = 0; j < 8; ++j) { float d = v[j] - mean; q += d * d; }
  for (int o = 16; o > 0; o >>= 1) q += __shfl_xor(q, o, 32);
  float rs = rsqrtf(q * (1.0f / 256.0f) + 1e-5f);
  bf16_t* o8 = out + (long)row * 256 + lane * 8;
#pragma unroll
  for (int j = 0; j < 8; ++j) {
    int col = lane * 8 + j;
    o8[j] = f2bf((v[j] - mean) * rs * gm[col] + bt[col]);
  }
}

// Softmax over rows of 64; one wave per row (2 elems/lane), bf16 output.
__global__ void kern_softmax(const float* __restrict__ logits, bf16_t* __restrict__ attn) {
  int row = blockIdx.x * 8 + (threadIdx.x >> 5);
  int lane = threadIdx.x & 31;
  long base = (long)row * 64 + lane * 2;
  float a = logits[base], b = logits[base + 1];
  float mx = fmaxf(a, b);
  for (int o = 16; o > 0; o >>= 1) mx = fmaxf(mx, __shfl_xor(mx, o, 32));
  float e0 = __expf(a - mx), e1 = __expf(b - mx);
  float s = e0 + e1;
  for (int o = 16; o > 0; o >>= 1) s += __shfl_xor(s, o, 32);
  float inv = 1.0f / s;
  attn[base]     = f2bf(e0 * inv);
  attn[base + 1] = f2bf(e1 * inv);
}

// ---------------------------------------------------------------------------
// LDS staging helpers (blockDim.x == 128)
// ---------------------------------------------------------------------------
// Straight copy of a 64x64 bf16 tile (contiguous source) into LDS.
__device__ __forceinline__ void stage_copy(bf16_t* dst, const bf16_t* src) {
  for (int e = threadIdx.x; e < 512; e += 128)
    *(v4u*)&dst[e * 8] = *(const v4u*)&src[e * 8];
}
// Copy 64 rows of 64 bf16 from strided global rows into LDS (row-major).
__device__ __forceinline__ void stage_rows(bf16_t* dst, const bf16_t* src, long row_stride) {
  for (int e = threadIdx.x; e < 512; e += 128) {
    int r = e >> 3, c = (e & 7) << 3;
    *(v4u*)&dst[r * 64 + c] = *(const v4u*)&src[(long)r * row_stride + c];
  }
}
// Transpose-stage: source is K x N row-major (64x64), dst is N x K row-major.
// 4x4 register-blocked: vector global reads + vector LDS writes.
__device__ __forceinline__ void stage_transpose(bf16_t* dst, const bf16_t* src,
                                                long row_stride) {
  for (int e = threadIdx.x; e < 256; e += 128) {
    int kb = (e >> 4) << 2, nb = (e & 15) << 2;
    __align__(8) bf16_t t[4][4];
#pragma unroll
    for (int i = 0; i < 4; ++i)
      *(unsigned long long*)t[i] =
          *(const unsigned long long*)&src[(long)(kb + i) * row_stride + nb];
#pragma unroll
    for (int j = 0; j < 4; ++j) {
      __align__(8) bf16_t w[4] = { t[0][j], t[1][j], t[2][j], t[3][j] };
      *(unsigned long long*)&dst[(nb + j) * 64 + kb] = *(unsigned long long*)w;
    }
  }
}
// Guarded variant (zero-fills columns beyond Nsize).
__device__ __forceinline__ void stage_transpose_guard(bf16_t* dst, const bf16_t* src,
                                                      long row_stride, int n0, int Nsize) {
  for (int e = threadIdx.x; e < 256; e += 128) {
    int kb = (e >> 4) << 2, nb = (e & 15) << 2;
    __align__(8) bf16_t t[4][4];
    bool ok = (n0 + nb + 4 <= Nsize);
#pragma unroll
    for (int i = 0; i < 4; ++i) {
      unsigned long long v = 0ull;
      if (ok) v = *(const unsigned long long*)&src[(long)(kb + i) * row_stride + nb];
      *(unsigned long long*)t[i] = v;
    }
#pragma unroll
    for (int j = 0; j < 4; ++j) {
      __align__(8) bf16_t w[4] = { t[0][j], t[1][j], t[2][j], t[3][j] };
      *(unsigned long long*)&dst[(nb + j) * 64 + kb] = *(unsigned long long*)w;
    }
  }
}

// Shared 64x64 compute step: 4 waves, wave w owns rows [16w,16w+16).
// As: 64xK(=64) row-major; Bt: 64(N) x 64(K) row-major (transposed operand).
__device__ __forceinline__ void mma_64x64(const bf16_t* As, const bf16_t* Bt, v8f acc[4]) {
  int w = threadIdx.x >> 5;
#pragma unroll
  for (int ka = 0; ka < 64; ka += 32) {
    v16bf va = frag_a(&As[w * 16 * 64 + ka], 64);
#pragma unroll
    for (int nt = 0; nt < 4; ++nt)
      acc[nt] = wmma_bf16(va, frag_bt(&Bt[(nt * 16) * 64 + ka], 64), acc[nt]);
  }
}

// ---------------------------------------------------------------------------
// Generic bf16 WMMA GEMM: C(MxN) = A(MxK) @ B(KxN) + bias, 64x64 tile/block.
// Writes bf16 (Cbf) or fp32 (Cf). M,K multiples of 64; N multiple of 32.
// ---------------------------------------------------------------------------
__global__ void kern_gemm(const bf16_t* __restrict__ A, int lda,
                          const bf16_t* __restrict__ Bm, int ldb,
                          bf16_t* Cbf, float* Cf, int ldc,
                          int Nsize, int Ksize, const float* __restrict__ bias) {
  __shared__ __align__(16) bf16_t As[64 * 64];
  __shared__ __align__(16) bf16_t Bs[64 * 64];
  int tid = threadIdx.x;
  int m0 = blockIdx.x * 64, n0 = blockIdx.y * 64;
  v8f acc[4] = {};
  for (int kk = 0; kk < Ksize; kk += 64) {
    stage_rows(As, &A[(long)m0 * lda + kk], lda);
    stage_transpose_guard(Bs, &Bm[(long)kk * ldb + n0], ldb, n0, Nsize);
    if (kk + 64 < Ksize) {
      __builtin_prefetch(&A[(long)(m0 + (tid >> 3)) * lda + kk + 64 + (tid & 7) * 8], 0, 1);
      __builtin_prefetch(&Bm[(long)(kk + 64 + (tid >> 3)) * ldb + n0], 0, 1);
    }
    __syncthreads();
    mma_64x64(As, Bs, acc);
    __syncthreads();
  }
  int w = tid >> 5, lane = tid & 31, nl = lane & 15, mb = (lane < 16) ? 0 : 8;
#pragma unroll
  for (int nt = 0; nt < 4; ++nt) {
    int n = n0 + nt * 16 + nl;
    if (n >= Nsize) continue;
    float bv = bias ? bias[n] : 0.0f;
#pragma unroll
    for (int j = 0; j < 8; ++j) {
      int m = m0 + w * 16 + mb + j;
      float v = acc[nt][j] + bv;
      if (Cbf) Cbf[(long)m * ldc + n] = f2bf(v);
      else     Cf [(long)m * ldc + n] = v;
    }
  }
}

// ---------------------------------------------------------------------------
// QKV: [Q|K|V] = x @ W*, scattered to (B,H,T,dh) bf16. grid (512,16,3), blk 128.
// ---------------------------------------------------------------------------
__global__ void kern_qkv(const bf16_t* __restrict__ X,
                         const bf16_t* __restrict__ wq, const bf16_t* __restrict__ wk,
                         const bf16_t* __restrict__ wv,
                         bf16_t* q, bf16_t* k, bf16_t* v) {
  const bf16_t* W = (blockIdx.z == 0) ? wq : (blockIdx.z == 1) ? wk : wv;
  bf16_t* D = (blockIdx.z == 0) ? q : (blockIdx.z == 1) ? k : v;
  __shared__ __align__(16) bf16_t As[64 * 64];
  __shared__ __align__(16) bf16_t Bs[64 * 64];
  int tid = threadIdx.x;
  int m0 = blockIdx.x * 64, n0 = blockIdx.y * 64;
  v8f acc[4] = {};
  for (int kk = 0; kk < 1024; kk += 64) {
    stage_rows(As, &X[(long)m0 * 1024 + kk], 1024);
    stage_transpose(Bs, &W[(long)kk * 1024 + n0], 1024);
    if (kk + 64 < 1024) {
      __builtin_prefetch(&X[(long)(m0 + (tid >> 3)) * 1024 + kk + 64 + (tid & 7) * 8], 0, 1);
      __builtin_prefetch(&W[(long)(kk + 64 + (tid >> 3)) * 1024 + n0], 0, 1);
    }
    __syncthreads();
    mma_64x64(As, Bs, acc);
    __syncthreads();
  }
  int w = tid >> 5, lane = tid & 31, nl = lane & 15, mb = (lane < 16) ? 0 : 8;
#pragma unroll
  for (int nt = 0; nt < 4; ++nt) {
    int n = n0 + nt * 16 + nl;
    int h = n >> 6, d = n & 63;
#pragma unroll
    for (int j = 0; j < 8; ++j) {
      int m = m0 + w * 16 + mb + j;
      int b = m >> 6, t = m & 63;
      D[(((long)b * 16 + h) * 64 + t) * 64 + d] = f2bf(acc[nt][j]);
    }
  }
}

// ---------------------------------------------------------------------------
// Smolgen final: logits = u2 @ shared_w + static_bias[h] + SCALE*aqak[h]
// M=8192 (b,h), N=4096 (t,s), K=256. grid (128,64), blk 128.
// ---------------------------------------------------------------------------
__global__ void kern_smol4(const bf16_t* __restrict__ U, const bf16_t* __restrict__ SW,
                           const float* __restrict__ sbias, const float* __restrict__ aqak,
                           float* __restrict__ logits) {
  __shared__ __align__(16) bf16_t As[64 * 64];
  __shared__ __align__(16) bf16_t Bs[64 * 64];
  int tid = threadIdx.x;
  int m0 = blockIdx.x * 64, n0 = blockIdx.y * 64;
  v8f acc[4] = {};
  for (int kk = 0; kk < 256; kk += 64) {
    stage_rows(As, &U[(long)m0 * 256 + kk], 256);
    stage_transpose(Bs, &SW[(long)kk * 4096 + n0], 4096);
    __syncthreads();
    mma_64x64(As, Bs, acc);
    __syncthreads();
  }
  int w = tid >> 5, lane = tid & 31, nl = lane & 15, mb = (lane < 16) ? 0 : 8;
#pragma unroll
  for (int nt = 0; nt < 4; ++nt) {
    int n = n0 + nt * 16 + nl;
#pragma unroll
    for (int j = 0; j < 8; ++j) {
      int m = m0 + w * 16 + mb + j;
      int h = m & 15;
      logits[(long)m * 4096 + n] =
          acc[nt][j] + sbias[(long)h * 4096 + n] + 0.125f * aqak[(long)h * 4096 + n];
    }
  }
}

// ---------------------------------------------------------------------------
// logits += SCALE * Q K^T per (b,h).  grid = B*H = 8192, blk 128.
// K tile is (s,d) = N x K already -> straight copy + frag_bt.
// ---------------------------------------------------------------------------
__global__ void kern_logits_qk(const bf16_t* __restrict__ Q, const bf16_t* __restrict__ K,
                               float* __restrict__ logits) {
  __shared__ __align__(16) bf16_t Qs[64 * 64];
  __shared__ __align__(16) bf16_t Ks[64 * 64];
  int tid = threadIdx.x;
  long bh = blockIdx.x;
  stage_copy(Qs, Q + bh * 4096);
  stage_copy(Ks, K + bh * 4096);
  __syncthreads();
  v8f acc[4] = {};
  mma_64x64(Qs, Ks, acc);
  int w = tid >> 5, lane = tid & 31, nl = lane & 15, mb = (lane < 16) ? 0 : 8;
  float* Lg = logits + bh * 4096;
#pragma unroll
  for (int nt = 0; nt < 4; ++nt)
#pragma unroll
    for (int j = 0; j < 8; ++j)
      Lg[(w * 16 + mb + j) * 64 + nt * 16 + nl] += 0.125f * acc[nt][j];
}

// ---------------------------------------------------------------------------
// logits[:,h,t,:] += SCALE * Q[:,h,t,:] @ aK[h,t]^T. grid (H*T=1024, B/64=8).
// aK[h,t] tile is (s,d) = N x K -> straight copy + frag_bt.
// ---------------------------------------------------------------------------
__global__ void kern_logits_ak(const bf16_t* __restrict__ Q, const bf16_t* __restrict__ aK,
                               float* __restrict__ logits) {
  __shared__ __align__(16) bf16_t As[64 * 64];
  __shared__ __align__(16) bf16_t Bs[64 * 64];
  int tid = threadIdx.x;
  int ht = blockIdx.x, h = ht >> 6, t = ht & 63;
  int b0 = blockIdx.y * 64;
  stage_rows(As, &Q[(((long)b0 * 16 + h) * 64 + t) * 64], 16 * 64 * 64);
  stage_copy(Bs, aK + (long)ht * 4096);
  __syncthreads();
  v8f acc[4] = {};
  mma_64x64(As, Bs, acc);
  int w = tid >> 5, lane = tid & 31, nl = lane & 15, mb = (lane < 16) ? 0 : 8;
#pragma unroll
  for (int nt = 0; nt < 4; ++nt)
#pragma unroll
    for (int j = 0; j < 8; ++j) {
      int bl = b0 + w * 16 + mb + j;
      logits[(((long)bl * 16 + h) * 64 + t) * 64 + nt * 16 + nl] += 0.125f * acc[nt][j];
    }
}

// ---------------------------------------------------------------------------
// logits[:,h,:,s] += SCALE * K[:,h,s,:] @ aQ[h,:,s,:]^T. grid (H*T=1024, 8).
// B rows (t) are contiguous in d -> row-strided copy + frag_bt.
// ---------------------------------------------------------------------------
__global__ void kern_logits_aq(const bf16_t* __restrict__ K, const bf16_t* __restrict__ aQ,
                               float* __restrict__ logits) {
  __shared__ __align__(16) bf16_t As[64 * 64];
  __shared__ __align__(16) bf16_t Bs[64 * 64];
  int tid = threadIdx.x;
  int hs = blockIdx.x, h = hs >> 6, s = hs & 63;
  int b0 = blockIdx.y * 64;
  stage_rows(As, &K[(((long)b0 * 16 + h) * 64 + s) * 64], 16 * 64 * 64);
  stage_rows(Bs, &aQ[(((long)h * 64) * 64 + s) * 64], 64 * 64);  // row t stride = 4096
  __syncthreads();
  v8f acc[4] = {};
  mma_64x64(As, Bs, acc);
  int w = tid >> 5, lane = tid & 31, nl = lane & 15, mb = (lane < 16) ? 0 : 8;
#pragma unroll
  for (int nt = 0; nt < 4; ++nt)
#pragma unroll
    for (int j = 0; j < 8; ++j) {
      int bl = b0 + w * 16 + mb + j;
      int t = nt * 16 + nl;
      logits[(((long)bl * 16 + h) * 64 + t) * 64 + s] += 0.125f * acc[nt][j];
    }
}

// ---------------------------------------------------------------------------
// out_pre[b,t,h*64+d] = attn[b,h] @ V[b,h].  grid = B*H, blk 128.
// V tile is (s,d) = K x N -> transpose-stage.
// ---------------------------------------------------------------------------
__global__ void kern_av(const bf16_t* __restrict__ attn, const bf16_t* __restrict__ V,
                        float* __restrict__ outp) {
  __shared__ __align__(16) bf16_t As[64 * 64];
  __shared__ __align__(16) bf16_t Bs[64 * 64];
  int tid = threadIdx.x;
  long bh = blockIdx.x;
  int b = blockIdx.x >> 4, h = blockIdx.x & 15;
  stage_copy(As, attn + bh * 4096);
  stage_transpose(Bs, V + bh * 4096, 64);
  __syncthreads();
  v8f acc[4] = {};
  mma_64x64(As, Bs, acc);
  int w = tid >> 5, lane = tid & 31, nl = lane & 15, mb = (lane < 16) ? 0 : 8;
#pragma unroll
  for (int nt = 0; nt < 4; ++nt)
#pragma unroll
    for (int j = 0; j < 8; ++j) {
      int t = w * 16 + mb + j, d = nt * 16 + nl;
      outp[((long)b * 64 + t) * 1024 + h * 64 + d] = acc[nt][j];
    }
}

// ---------------------------------------------------------------------------
// out_pre[b,t,h*64+d] += attn[:,h,t,:] @ aV[h,t].  grid (H*T=1024, 8).
// aV[h,t] tile is (s,d) = K x N -> transpose-stage.
// ---------------------------------------------------------------------------
__global__ void kern_aav(const bf16_t* __restrict__ attn, const bf16_t* __restrict__ aV,
                         float* __restrict__ outp) {
  __shared__ __align__(16) bf16_t As[64 * 64];
  __shared__ __align__(16) bf16_t Bs[64 * 64];
  int tid = threadIdx.x;
  int ht = blockIdx.x, h = ht >> 6, t = ht & 63;
  int b0 = blockIdx.y * 64;
  stage_rows(As, &attn[(((long)b0 * 16 + h) * 64 + t) * 64], 16 * 64 * 64);
  stage_transpose(Bs, aV + (long)ht * 4096, 64);
  __syncthreads();
  v8f acc[4] = {};
  mma_64x64(As, Bs, acc);
  int w = tid >> 5, lane = tid & 31, nl = lane & 15, mb = (lane < 16) ? 0 : 8;
#pragma unroll
  for (int nt = 0; nt < 4; ++nt)
#pragma unroll
    for (int j = 0; j < 8; ++j) {
      int bl = b0 + w * 16 + mb + j, d = nt * 16 + nl;
      outp[((long)bl * 64 + t) * 1024 + h * 64 + d] += acc[nt][j];
    }
}

// ---------------------------------------------------------------------------
// Final: y = (out_pre @ Wo + b) * RES_SCALE. A staged fp32->bf16. grid (512,16).
// ---------------------------------------------------------------------------
__global__ void kern_out(const float* __restrict__ Ap, const bf16_t* __restrict__ W,
                         const float* __restrict__ bias, float* __restrict__ Y) {
  __shared__ __align__(16) bf16_t As[64 * 64];
  __shared__ __align__(16) bf16_t Bs[64 * 64];
  int tid = threadIdx.x;
  int m0 = blockIdx.x * 64, n0 = blockIdx.y * 64;
  v8f acc[4] = {};
  for (int kk = 0; kk < 1024; kk += 64) {
    for (int e = tid; e < 512; e += 128) {
      int r = e >> 3, c = (e & 7) << 3;
      const float* src = &Ap[(long)(m0 + r) * 1024 + kk + c];
      float4 v0 = *(const float4*)src;
      float4 v1 = *(const float4*)(src + 4);
      __align__(16) bf16_t w8[8] = { f2bf(v0.x), f2bf(v0.y), f2bf(v0.z), f2bf(v0.w),
                                     f2bf(v1.x), f2bf(v1.y), f2bf(v1.z), f2bf(v1.w) };
      *(v4u*)&As[r * 64 + c] = *(v4u*)w8;
    }
    stage_transpose(Bs, &W[(long)kk * 1024 + n0], 1024);
    if (kk + 64 < 1024) {
      __builtin_prefetch(&Ap[(long)(m0 + (tid >> 3)) * 1024 + kk + 64 + (tid & 7) * 8], 0, 1);
      __builtin_prefetch(&W[(long)(kk + 64 + (tid >> 3)) * 1024 + n0], 0, 1);
    }
    __syncthreads();
    mma_64x64(As, Bs, acc);
    __syncthreads();
  }
  int w = tid >> 5, lane = tid & 31, nl = lane & 15, mb = (lane < 16) ? 0 : 8;
#pragma unroll
  for (int nt = 0; nt < 4; ++nt) {
    int n = n0 + nt * 16 + nl;
#pragma unroll
    for (int j = 0; j < 8; ++j) {
      int m = m0 + w * 16 + mb + j;
      Y[(long)m * 1024 + n] = (acc[nt][j] + bias[n]) * 0.70710678118654752f;
    }
  }
}

// ---------------------------------------------------------------------------
// Host launcher
// ---------------------------------------------------------------------------
extern "C" void kernel_launch(void* const* d_in, const int* in_sizes, int n_in,
                              void* d_out, int out_size, void* d_ws, size_t ws_size,
                              hipStream_t stream) {
  (void)in_sizes; (void)n_in; (void)out_size; (void)ws_size;
  const float* x        = (const float*)d_in[0];
  const float* Wq       = (const float*)d_in[1];
  const float* Wk       = (const float*)d_in[2];
  const float* Wv       = (const float*)d_in[3];
  const float* Wo_w     = (const float*)d_in[4];
  const float* Wo_bias  = (const float*)d_in[5];
  const float* sbias    = (const float*)d_in[6];
  const float* aQ       = (const float*)d_in[7];
  const float* aK       = (const float*)d_in[8];
  const float* aV       = (const float*)d_in[9];
  const float* pt_w     = (const float*)d_in[10];
  const float* pt_b     = (const float*)d_in[11];
  const float* pg_w     = (const float*)d_in[12];
  const float* pg_b     = (const float*)d_in[13];
  const float* ph_w     = (const float*)d_in[14];
  const float* ph_b     = (const float*)d_in[15];
  const float* ln1_g    = (const float*)d_in[16];
  const float* ln1_b    = (const float*)d_in[17];
  const float* ln2_g    = (const float*)d_in[18];
  const float* ln2_b    = (const float*)d_in[19];
  const float* shared_w = (const float*)d_in[20];
  float* y = (float*)d_out;

  char* ws = (char*)d_ws;
  size_t off = 0;
  auto alloc = [&](size_t bytes) -> void* {
    void* p = ws + off;
    off = (off + bytes + 255) & ~(size_t)255;
    return p;
  };
  const long NX = 512L * 64 * 1024;          // 33.5M
  bf16_t* xb    = (bf16_t*)alloc(NX * 2);
  bf16_t* wqb   = (bf16_t*)alloc(1048576L * 2);
  bf16_t* wkb   = (bf16_t*)alloc(1048576L * 2);
  bf16_t* wvb   = (bf16_t*)alloc(1048576L * 2);
  bf16_t* wob   = (bf16_t*)alloc(1048576L * 2);
  bf16_t* aQb   = (bf16_t*)alloc(4194304L * 2);
  bf16_t* aKb   = (bf16_t*)alloc(4194304L * 2);
  bf16_t* aVb   = (bf16_t*)alloc(4194304L * 2);
  bf16_t* ptwb  = (bf16_t*)alloc(32768L * 2);
  bf16_t* pgwb  = (bf16_t*)alloc(524288L * 2);
  bf16_t* phwb  = (bf16_t*)alloc(1048576L * 2);
  bf16_t* shwb  = (bf16_t*)alloc(1048576L * 2);
  bf16_t* Qb    = (bf16_t*)alloc(NX * 2);     // (B,H,T,dh)
  bf16_t* Kb    = (bf16_t*)alloc(NX * 2);
  bf16_t* Vb    = (bf16_t*)alloc(NX * 2);
  bf16_t* zbf   = (bf16_t*)alloc(1048576L * 2);   // (B, T*CD)
  float*  gf    = (float*) alloc(131072L * 4);    // (B, SD)
  bf16_t* gbf   = (bf16_t*)alloc(131072L * 2);
  float*  uf    = (float*) alloc(2097152L * 4);   // (B, H*SD)
  bf16_t* ubf   = (bf16_t*)alloc(2097152L * 2);
  float*  aqak  = (float*) alloc(65536L * 4);     // (H,T,T)
  float*  logits= (float*) alloc(33554432L * 4);  // (B,H,T,T)
  bf16_t* attnb = (bf16_t*)alloc(33554432L * 2);
  float*  outp  = (float*) alloc(NX * 4);         // (B,T,Dm)

  auto pk = [&](const float* s, bf16_t* d, long n) {
    kern_pack<<<dim3((unsigned)((n + 255) / 256)), 256, 0, stream>>>(s, d, n);
  };
  pk(x, xb, NX);
  pk(Wq, wqb, 1048576); pk(Wk, wkb, 1048576); pk(Wv, wvb, 1048576); pk(Wo_w, wob, 1048576);
  pk(aQ, aQb, 4194304); pk(aK, aKb, 4194304); pk(aV, aVb, 4194304);
  pk(pt_w, ptwb, 32768); pk(pg_w, pgwb, 524288); pk(ph_w, phwb, 1048576);
  pk(shared_w, shwb, 1048576);

  kern_aqak<<<dim3(256), 256, 0, stream>>>(aQ, aK, aqak);

  // Q,K,V projections -> (B,H,T,dh) bf16
  kern_qkv<<<dim3(512, 16, 3), 128, 0, stream>>>(xb, wqb, wkb, wvb, Qb, Kb, Vb);

  // Smolgen chain
  kern_gemm<<<dim3(512, 1), 128, 0, stream>>>(xb, 1024, ptwb, 32, zbf, nullptr, 32,
                                              32, 1024, pt_b);
  kern_gemm<<<dim3(8, 4), 128, 0, stream>>>(zbf, 2048, pgwb, 256, nullptr, gf, 256,
                                            256, 2048, pg_b);
  kern_silu_ln<<<dim3(128), 128, 0, stream>>>(gf, gbf, ln1_g, ln1_b);
  kern_gemm<<<dim3(8, 64), 128, 0, stream>>>(gbf, 256, phwb, 4096, nullptr, uf, 4096,
                                             4096, 256, ph_b);
  kern_silu_ln<<<dim3(2048), 128, 0, stream>>>(uf, ubf, ln2_g, ln2_b);
  // logits = smolgen + static_bias + SCALE*aQaK (base write)
  kern_smol4<<<dim3(128, 64), 128, 0, stream>>>(ubf, shwb, sbias, aqak, logits);

  // logits accumulation
  kern_logits_qk<<<dim3(8192), 128, 0, stream>>>(Qb, Kb, logits);
  kern_logits_ak<<<dim3(1024, 8), 128, 0, stream>>>(Qb, aKb, logits);
  kern_logits_aq<<<dim3(1024, 8), 128, 0, stream>>>(Kb, aQb, logits);

  // softmax -> bf16 attention
  kern_softmax<<<dim3(65536), 256, 0, stream>>>(logits, attnb);

  // attention output
  kern_av <<<dim3(8192), 128, 0, stream>>>(attnb, Vb, outp);
  kern_aav<<<dim3(1024, 8), 128, 0, stream>>>(attnb, aVb, outp);

  // final projection
  kern_out<<<dim3(512, 16), 128, 0, stream>>>(outp, wob, Wo_bias, y);
}
